// MultiHeadAttention_23553600651289
// MI455X (gfx1250) — matrix operand order repaired and checked by reading, not compiled, not run on hardware
//
#include <hip/hip_runtime.h>
#include <hip/hip_bf16.h>

// ---------------------------------------------------------------------------
// MHA forward, CDNA5 (gfx1250), bf16 WMMA with fp32 accumulate.
// B=4, S=2048, D=768, H=12, HD=64.
// GEMMs: ping-pong fragment buffers (unroll-by-2) -> no register copies,
// loads issued a half-iteration ahead of the WMMAs that consume them.
// ---------------------------------------------------------------------------

typedef __bf16 bf16_t;
typedef __attribute__((ext_vector_type(16))) __bf16 v16bf;
typedef __attribute__((ext_vector_type(8)))  __bf16 v8bf;
typedef __attribute__((ext_vector_type(8)))  float  v8f;

#define BB 4
#define SS 2048
#define DD 768
#define HH 12
#define HD 64

// A-fragment (16x32 bf16, MxK): lane m<16 holds K {koff..koff+15}
// with koff = 16*(lane>=16). Caller passes p = row_base + k_base + koff.
static __device__ __forceinline__ v16bf ld_fragA(const bf16_t* p) {
  v8bf lo = *reinterpret_cast<const v8bf*>(p);
  v8bf hi = *reinterpret_cast<const v8bf*>(p + 8);
  v16bf r;
#pragma unroll
  for (int i = 0; i < 8; ++i) { r[i] = lo[i]; r[i + 8] = hi[i]; }
  return r;
}

// B-fragment (32x16 bf16, KxN): lane n holds column (n&15), 16 contiguous K
// values starting at K = 16*(lane>=16). Caller passes p = col_base + k_base
// + 16*(lane>=16).
static __device__ __forceinline__ v16bf ld_fragB(const bf16_t* p) {
  v8bf lo = *reinterpret_cast<const v8bf*>(p);
  v8bf hi = *reinterpret_cast<const v8bf*>(p + 8);
  v16bf r;
#pragma unroll
  for (int i = 0; i < 8; ++i) { r[i] = lo[i]; r[i + 8] = hi[i]; }
  return r;
}

static __device__ __forceinline__ v8f wmma_bf16(v16bf a, v16bf b, v8f c) {
  return __builtin_amdgcn_wmma_f32_16x16x32_bf16(false, a, false, b,
                                                 (short)0, c, false, false);
}

// ---------------------------------------------------------------------------
// Kernel 1: fp32 -> bf16 conversion (grid-stride).
// ---------------------------------------------------------------------------
__global__ void f32_to_bf16(const float* __restrict__ s, bf16_t* __restrict__ d, int n) {
  int i = blockIdx.x * blockDim.x + threadIdx.x;
  int stride = gridDim.x * blockDim.x;
  for (; i < n; i += stride) d[i] = (bf16_t)s[i];
}

// ---------------------------------------------------------------------------
// Kernel 2: fused QKV projection. y = x @ W^T.
// One wave per 16(M)x64(N) tile per projection; ping-pong fragments.
// Q,K -> [B,H,S,HD]; V -> transposed [B,H,HD,S].
// ---------------------------------------------------------------------------
__global__ __launch_bounds__(256) void qkv_gemm(
    const bf16_t* __restrict__ xb,
    const bf16_t* __restrict__ wqb, const bf16_t* __restrict__ wkb,
    const bf16_t* __restrict__ wvb,
    bf16_t* __restrict__ Qb, bf16_t* __restrict__ Kb, bf16_t* __restrict__ vTb) {
  const int wid = (blockIdx.x * blockDim.x + threadIdx.x) >> 5;
  const int M_TILES = (BB * SS) / 16;   // 512
  const int N64 = DD / 64;              // 12
  if (wid >= 3 * M_TILES * N64) return;
  const int proj = wid / (M_TILES * N64);
  const int rem  = wid % (M_TILES * N64);
  const int mt = rem / N64, n64 = rem % N64;
  const int lane = threadIdx.x & 31;
  const int lhalf = lane >> 4;
  const int ln = lane & 15;
  const int koff = lhalf * 16;

  const bf16_t* w = (proj == 0) ? wqb : ((proj == 1) ? wkb : wvb);
  const bf16_t* arow = xb + (size_t)(mt * 16 + ln) * DD + koff;
  const bf16_t* brow[4];
#pragma unroll
  for (int t = 0; t < 4; ++t)
    brow[t] = w + (size_t)(n64 * 64 + t * 16 + ln) * DD + lhalf * 16;

  v8f acc[4];
#pragma unroll
  for (int t = 0; t < 4; ++t)
#pragma unroll
    for (int i = 0; i < 8; ++i) acc[t][i] = 0.0f;

  // Ping buffers preloaded at k=0. DD % 64 == 0.
  v16bf a0 = ld_fragA(arow);
  v16bf b0[4];
#pragma unroll
  for (int t = 0; t < 4; ++t) b0[t] = ld_fragB(brow[t]);

  for (int k = 0; k < DD; k += 64) {
    // Pong buffers (fresh per iteration -> no cross-iteration copies).
    v16bf a1 = ld_fragA(arow + (k + 32));
    v16bf b1[4];
#pragma unroll
    for (int t = 0; t < 4; ++t) b1[t] = ld_fragB(brow[t] + (k + 32));

#pragma unroll
    for (int t = 0; t < 4; ++t) acc[t] = wmma_bf16(a0, b0[t], acc[t]);

    if (k + 64 < DD) {   // reload ping in place; WAR safe under in-order issue
      a0 = ld_fragA(arow + (k + 64));
#pragma unroll
      for (int t = 0; t < 4; ++t) b0[t] = ld_fragB(brow[t] + (k + 64));
    }

#pragma unroll
    for (int t = 0; t < 4; ++t) acc[t] = wmma_bf16(a1, b1[t], acc[t]);
  }

  const int m0 = mt * 16;
#pragma unroll
  for (int t = 0; t < 4; ++t) {
    const int hd = t * 16 + ln;
#pragma unroll
    for (int i = 0; i < 8; ++i) {
      const int m = m0 + i + lhalf * 8;
      const int b = m >> 11;          // / SS
      const int s = m & (SS - 1);
      const bf16_t v = (bf16_t)acc[t][i];
      if (proj == 2)
        vTb[(((size_t)b * HH + n64) * HD + hd) * SS + s] = v;
      else if (proj == 0)
        Qb[(((size_t)b * HH + n64) * SS + s) * HD + hd] = v;
      else
        Kb[(((size_t)b * HH + n64) * SS + s) * HD + hd] = v;
    }
  }
}

// ---------------------------------------------------------------------------
// Kernel 3: causal flash attention. One wave per (b, h, 16-query tile).
// Online softmax; K fragments reloaded in place one key-tile ahead; V loads
// hoisted above the softmax VALU block so global latency overlaps exp work.
// ---------------------------------------------------------------------------
__global__ __launch_bounds__(128) void attn(
    const bf16_t* __restrict__ Qb, const bf16_t* __restrict__ Kb,
    const bf16_t* __restrict__ vTb, bf16_t* __restrict__ ctxb) {
  __shared__ bf16_t Pls[4][16][32];   // 1 KB per wave

  const int wid = (blockIdx.x * blockDim.x + threadIdx.x) >> 5;
  const int QT = SS / 16;             // 128
  if (wid >= BB * HH * QT) return;
  const int qt = wid % QT;
  const int bh = wid / QT;
  const int h = bh % HH, b = bh / HH;
  const int q0 = qt * 16;

  const int lane = threadIdx.x & 31;
  const int lhalf = lane >> 4;
  const int ln = lane & 15;
  const int koff = lhalf * 16;
  bf16_t (*P)[32] = Pls[threadIdx.x >> 5];

  const bf16_t* Qp = Qb  + ((size_t)(b * HH + h) * SS) * HD;
  const bf16_t* Kp = Kb  + ((size_t)(b * HH + h) * SS) * HD + lhalf * 16;
  const bf16_t* Vt = vTb + ((size_t)(b * HH + h) * HD) * SS;

  // Q fragments (HD=64 -> two K-steps), register-resident for the whole loop.
  const v16bf qf0 = ld_fragA(Qp + (size_t)(q0 + ln) * HD + 0  + koff);
  const v16bf qf1 = ld_fragA(Qp + (size_t)(q0 + ln) * HD + 32 + koff);

  float mrow[8], lrow[8];
  v8f o[4];
#pragma unroll
  for (int i = 0; i < 8; ++i) { mrow[i] = -1.0e30f; lrow[i] = 0.0f; }
#pragma unroll
  for (int t = 0; t < 4; ++t)
#pragma unroll
    for (int i = 0; i < 8; ++i) o[t][i] = 0.0f;

  const float scale = 0.125f;         // 1/sqrt(64)
  const int kend = q0 + 16;           // causal: keys <= q0+15

  // Preload K fragments for the first key tile.
  v16bf kfa0 = ld_fragB(Kp + (size_t)(0 + ln) * HD + 0);
  v16bf kfa1 = ld_fragB(Kp + (size_t)(0 + ln) * HD + 32);
  v16bf kfb0 = ld_fragB(Kp + (size_t)(16 + ln) * HD + 0);
  v16bf kfb1 = ld_fragB(Kp + (size_t)(16 + ln) * HD + 32);

  for (int k0 = 0; k0 < kend; k0 += 32) {
    // ---- scores: 16 queries x 32 keys via 4 WMMAs ----
    v8f s0, s1;
#pragma unroll
    for (int i = 0; i < 8; ++i) { s0[i] = 0.0f; s1[i] = 0.0f; }
    s0 = wmma_bf16(qf0, kfa0, s0);
    s0 = wmma_bf16(qf1, kfa1, s0);
    s1 = wmma_bf16(qf0, kfb0, s1);
    s1 = wmma_bf16(qf1, kfb1, s1);

    // ---- V fragments for THIS tile: independent of softmax, issue early ----
    v16bf vf[4];
#pragma unroll
    for (int t = 0; t < 4; ++t)
      vf[t] = ld_fragB(Vt + (size_t)(t * 16 + ln) * SS + k0 + lhalf * 16);

    // ---- K fragments for NEXT tile: reload in place (after consumption) ----
    const int k1 = k0 + 32;
    if (k1 < kend) {
      kfa0 = ld_fragB(Kp + (size_t)(k1 + ln) * HD + 0);
      kfa1 = ld_fragB(Kp + (size_t)(k1 + ln) * HD + 32);
      kfb0 = ld_fragB(Kp + (size_t)(k1 + 16 + ln) * HD + 0);
      kfb1 = ld_fragB(Kp + (size_t)(k1 + 16 + ln) * HD + 32);
      __builtin_prefetch(Vt + (size_t)ln * SS + k1, 0, 1);
    }

    // ---- causal mask, scale, online softmax (per-row over 16 lanes) ----
    float corr[8];
#pragma unroll
    for (int i = 0; i < 8; ++i) {
      const int row = q0 + i + lhalf * 8;
      float a0 = s0[i] * scale;
      float a1 = s1[i] * scale;
      if (k0 + ln      > row) a0 = -3.0e38f;
      if (k0 + 16 + ln > row) a1 = -3.0e38f;
      float tm = fmaxf(a0, a1);
      tm = fmaxf(tm, __shfl_xor(tm, 1, 32));
      tm = fmaxf(tm, __shfl_xor(tm, 2, 32));
      tm = fmaxf(tm, __shfl_xor(tm, 4, 32));
      tm = fmaxf(tm, __shfl_xor(tm, 8, 32));
      const float mnew = fmaxf(mrow[i], tm);
      corr[i] = __expf(mrow[i] - mnew);
      const float p0 = __expf(a0 - mnew);
      const float p1 = __expf(a1 - mnew);
      float rs = p0 + p1;
      rs += __shfl_xor(rs, 1, 32);
      rs += __shfl_xor(rs, 2, 32);
      rs += __shfl_xor(rs, 4, 32);
      rs += __shfl_xor(rs, 8, 32);
      lrow[i] = lrow[i] * corr[i] + rs;
      mrow[i] = mnew;
      // C-layout -> LDS (row-major 16x32 P tile) as bf16
      P[i + lhalf * 8][ln]      = (bf16_t)p0;
      P[i + lhalf * 8][16 + ln] = (bf16_t)p1;
    }
#pragma unroll
    for (int t = 0; t < 4; ++t)
#pragma unroll
      for (int i = 0; i < 8; ++i) o[t][i] *= corr[i];

    // ---- P (16x32) @ V (32x64): reload P in A-fragment layout, 4 WMMAs ----
    v16bf pf = ld_fragA(&P[ln][0] + koff);
#pragma unroll
    for (int t = 0; t < 4; ++t) o[t] = wmma_bf16(pf, vf[t], o[t]);
  }

  // ---- epilogue: normalize and scatter ctx[b, s, h*64 + d] as bf16 ----
#pragma unroll
  for (int t = 0; t < 4; ++t) {
    const int col = h * HD + t * 16 + ln;
#pragma unroll
    for (int i = 0; i < 8; ++i) {
      const int row = q0 + i + lhalf * 8;
      ctxb[((size_t)b * SS + row) * DD + col] = (bf16_t)(o[t][i] / lrow[i]);
    }
  }
}

// ---------------------------------------------------------------------------
// Kernel 4: output projection. out = ctx @ wo^T + bo, fp32 stores.
// Same ping-pong pipeline as kernel 2.
// ---------------------------------------------------------------------------
__global__ __launch_bounds__(256) void out_gemm(
    const bf16_t* __restrict__ ctxb, const bf16_t* __restrict__ wob,
    const float* __restrict__ bo, float* __restrict__ out) {
  const int wid = (blockIdx.x * blockDim.x + threadIdx.x) >> 5;
  const int M_TILES = (BB * SS) / 16;
  const int N64 = DD / 64;
  if (wid >= M_TILES * N64) return;
  const int mt = wid / N64, n64 = wid % N64;
  const int lane = threadIdx.x & 31;
  const int lhalf = lane >> 4;
  const int ln = lane & 15;
  const int koff = lhalf * 16;

  const bf16_t* arow = ctxb + (size_t)(mt * 16 + ln) * DD + koff;
  const bf16_t* brow[4];
#pragma unroll
  for (int t = 0; t < 4; ++t)
    brow[t] = wob + (size_t)(n64 * 64 + t * 16 + ln) * DD + lhalf * 16;

  v8f acc[4];
#pragma unroll
  for (int t = 0; t < 4; ++t)
#pragma unroll
    for (int i = 0; i < 8; ++i) acc[t][i] = 0.0f;

  v16bf a0 = ld_fragA(arow);
  v16bf b0[4];
#pragma unroll
  for (int t = 0; t < 4; ++t) b0[t] = ld_fragB(brow[t]);

  for (int k = 0; k < DD; k += 64) {
    v16bf a1 = ld_fragA(arow + (k + 32));
    v16bf b1[4];
#pragma unroll
    for (int t = 0; t < 4; ++t) b1[t] = ld_fragB(brow[t] + (k + 32));

#pragma unroll
    for (int t = 0; t < 4; ++t) acc[t] = wmma_bf16(a0, b0[t], acc[t]);

    if (k + 64 < DD) {
      a0 = ld_fragA(arow + (k + 64));
#pragma unroll
      for (int t = 0; t < 4; ++t) b0[t] = ld_fragB(brow[t] + (k + 64));
    }

#pragma unroll
    for (int t = 0; t < 4; ++t) acc[t] = wmma_bf16(a1, b1[t], acc[t]);
  }

  const int m0 = mt * 16;
#pragma unroll
  for (int t = 0; t < 4; ++t) {
    const int col = n64 * 64 + t * 16 + ln;
    const float bias = bo[col];
#pragma unroll
    for (int i = 0; i < 8; ++i) {
      const int m = m0 + i + lhalf * 8;
      out[(size_t)m * DD + col] = acc[t][i] + bias;
    }
  }
}

// ---------------------------------------------------------------------------
extern "C" void kernel_launch(void* const* d_in, const int* in_sizes, int n_in,
                              void* d_out, int out_size, void* d_ws, size_t ws_size,
                              hipStream_t stream) {
  (void)in_sizes; (void)n_in; (void)out_size; (void)ws_size;
  const float* x  = (const float*)d_in[0];
  const float* wq = (const float*)d_in[1];
  const float* wk = (const float*)d_in[2];
  const float* wv = (const float*)d_in[3];
  const float* wo = (const float*)d_in[4];
  const float* bo = (const float*)d_in[5];

  char* ws = (char*)d_ws;
  size_t off = 0;
  auto take = [&](size_t bytes) -> char* {
    char* p = ws + off;
    off = (off + bytes + 255) & ~(size_t)255;
    return p;
  };
  bf16_t* xb   = (bf16_t*)take((size_t)BB * SS * DD * 2);
  bf16_t* wqb  = (bf16_t*)take((size_t)DD * DD * 2);
  bf16_t* wkb  = (bf16_t*)take((size_t)DD * DD * 2);
  bf16_t* wvb  = (bf16_t*)take((size_t)DD * DD * 2);
  bf16_t* wob  = (bf16_t*)take((size_t)DD * DD * 2);
  bf16_t* Qb   = (bf16_t*)take((size_t)BB * HH * SS * HD * 2);
  bf16_t* Kb   = (bf16_t*)take((size_t)BB * HH * SS * HD * 2);
  bf16_t* vTb  = (bf16_t*)take((size_t)BB * HH * HD * SS * 2);
  bf16_t* ctxb = (bf16_t*)take((size_t)BB * SS * DD * 2);

  f32_to_bf16<<<2048, 256, 0, stream>>>(x,  xb,  BB * SS * DD);
  f32_to_bf16<<<512,  256, 0, stream>>>(wq, wqb, DD * DD);
  f32_to_bf16<<<512,  256, 0, stream>>>(wk, wkb, DD * DD);
  f32_to_bf16<<<512,  256, 0, stream>>>(wv, wvb, DD * DD);
  f32_to_bf16<<<512,  256, 0, stream>>>(wo, wob, DD * DD);

  const int gemm_waves = 3 * ((BB * SS) / 16) * (DD / 64);   // 18432
  qkv_gemm<<<(gemm_waves * 32 + 255) / 256, 256, 0, stream>>>(
      xb, wqb, wkb, wvb, Qb, Kb, vTb);

  const int attn_waves = BB * HH * (SS / 16);                // 6144
  attn<<<(attn_waves * 32 + 127) / 128, 128, 0, stream>>>(Qb, Kb, vTb, ctxb);

  const int op_waves = ((BB * SS) / 16) * (DD / 64);         // 6144
  out_gemm<<<(op_waves * 32 + 255) / 256, 256, 0, stream>>>(
      ctxb, wob, bo, (float*)d_out);
}
